// PolymerJEPAv2_65000035058119
// MI455X (gfx1250) — compile-verified, matrix operand order checked by reading, not run on hardware
//
#include <hip/hip_runtime.h>
#include <hip/hip_bf16.h>

// ---------------------------------------------------------------------------
// PolymerJEPAv2 pipeline for gfx1250 (MI455X).
// Dense GEMMs run on v_wmma_f32_16x16x32_bf16 (bf16 A/B staged in LDS, f32 acc).
// Tile fills use float4 global loads + packed bf16 ds_store_b64 on the aligned
// fast path (K=256 layers, edge-MLP body); uniform fullM/ragged branch keeps
// per-element exec masking out of the common path. Scatter phases (segment
// sums over 600K/400K edges x 256 feats) are HBM-bound; they use wave-per-row
// coalesced atomics. Everything launches on `stream`, scratch only in
// d_ws / d_out (graph-capture safe: kernels + hipMemsetAsync).
// ---------------------------------------------------------------------------

#define NHID   256
#define NFN    133
#define NFE    14
#define RWD    20
#define PRWD   20
#define NNODE  100000
#define NEDGE  400000
#define NSUBN  200000
#define NSUBE  600000
#define GB     256
#define NPATCH 2048
#define NTGT   4

typedef __attribute__((ext_vector_type(16))) __bf16 v16bf;
typedef __attribute__((ext_vector_type(4)))  __bf16 bf16x4;
typedef __attribute__((ext_vector_type(8)))  float  v8f;
typedef __attribute__((ext_vector_type(4)))  float  f32x4;

// GEMM tiling: 128x64 output tile per 256-thread (8-wave) workgroup,
// K staged in 32-wide bf16 slices.  LDS row strides padded (40 halves = 80B)
// so the 16B fragment chunks stay 16B-aligned and bank groups are spread.
#define MB   128
#define NBLK 64
#define KB   32
#define LDSA 40
#define LDSB 40

// A-operand fetch modes (template parameter -> specialized fill loops):
//   0: plain row-major A (optional per-row scale)
//   2: edge rows, h[idx0[row]] for k<256, edge_attr[idx1[row]] tail (ctx branch)
//   3: edge rows, h[idx0[row]] for k<256, edge_attr[row] tail (tgt branch)
struct GemmA {
  const float* A;   int lda;    // primary matrix
  const int*   idx0;            // row gather (edge src) for modes 2/3
  const int*   idx1;            // edge->attr mapper (mode 2)
  const float* ea;              // edge_attr
  const float* row_scale;       // optional per-row scale (mode 0)
  int K1;                       // split point (modes 2/3)
  int vec4;                     // mode 0: rows 16B-aligned AND K % 4 == 0
};

template <int MODE>
__device__ __forceinline__ float fetchA(const GemmA& g, int row, int k, int K) {
  if (MODE == 0) {
    float v = (k < K) ? g.A[(long long)row * g.lda + k] : 0.f;
    if (g.row_scale) v *= g.row_scale[row];
    return v;
  } else {
    if (k < g.K1) {
      return g.A[(long long)g.idx0[row] * g.lda + k];
    } else if (k < K) {
      int er = (MODE == 2) ? g.idx1[row] : row;
      return g.ea[(long long)er * NFE + (k - g.K1)];
    }
    return 0.f;
  }
}

// One f32x4 -> bf16x4 quad into LDS (8B ds_store_b64, offsets stay 8B-aligned)
__device__ __forceinline__ void store_quad(__bf16* dst, f32x4 v) {
  bf16x4 q;
#pragma unroll
  for (int c = 0; c < 4; ++c) q[c] = (__bf16)v[c];
  *(bf16x4*)dst = q;
}

// CDNA5 16-bit operand fragment (ISA 7.12.2): lanes 0-15 -> row=lane,
// K = {kh*8..kh*8+7, 16+kh*8..16+kh*8+7} with kh = lane>>4.  Both A (row-major
// strip) and B (staged N-major in LDS) use this mapping.
__device__ __forceinline__ v16bf frag_from_lds(const __bf16* b, int stride, int lane) {
  const int r = lane & 15, kh = lane >> 4;
  const __bf16* p = b + r * stride;
  v16bf f;
#pragma unroll
  for (int i = 0; i < 8; ++i) f[i]     = p[kh * 8 + i];
#pragma unroll
  for (int i = 0; i < 8; ++i) f[8 + i] = p[16 + kh * 8 + i];
  return f;
}

// flags: bit0 = ReLU, bit1 = add residual.  C width is always NHID.
template <int MODE>
__global__ __launch_bounds__(256)
void wmma_gemm(GemmA ga, const float* __restrict__ B, const float* __restrict__ bias,
               const float* __restrict__ resid, float* __restrict__ C,
               int M, int K, int flags) {
  __shared__ __align__(16) __bf16 As[MB * LDSA];
  __shared__ __align__(16) __bf16 Bs[NBLK * LDSB];
  const int tid  = threadIdx.x;
  const int lane = tid & 31;
  const int wave = tid >> 5;
  const int m0   = blockIdx.x * MB;
  const int n0   = blockIdx.y * NBLK;
  const int KT   = (K + KB - 1) / KB;
  const bool fullM = (m0 + MB <= M);     // uniform: interior M-block

  v8f acc[4];
#pragma unroll
  for (int t = 0; t < 4; ++t)
#pragma unroll
    for (int j = 0; j < 8; ++j) acc[t][j] = 0.f;

  for (int kt = 0; kt < KT; ++kt) {
    const int kbase = kt * KB;
    __syncthreads();

    // ---- A tile 128x32 (f32 -> bf16) ----------------------------------
    // Vec path (full quads only, no per-element guards when fullM):
    //   mode 0 with vec4 (K multiple of KB), or the k<K1 body of edge modes.
    const bool vec = (MODE == 0) ? (ga.vec4 != 0 && kbase + KB <= K)
                                 : (kbase + KB <= ga.K1);
    if (vec) {
      if (fullM) {
#pragma unroll
        for (int t = 0; t < (MB * KB) / (256 * 4); ++t) {
          int e4 = tid + t * 256;
          int m  = e4 >> 3;
          int kq = (e4 & 7) << 2;
          int gr = m0 + m;
          long long row = (MODE == 0) ? gr : ga.idx0[gr];
          f32x4 v = *(const f32x4*)(ga.A + row * ga.lda + kbase + kq);
          if (MODE == 0 && ga.row_scale) {
            float s = ga.row_scale[gr];
#pragma unroll
            for (int c = 0; c < 4; ++c) v[c] *= s;
          }
          store_quad(&As[m * LDSA + kq], v);
        }
      } else {
#pragma unroll
        for (int t = 0; t < (MB * KB) / (256 * 4); ++t) {
          int e4 = tid + t * 256;
          int m  = e4 >> 3;
          int kq = (e4 & 7) << 2;
          int gr = m0 + m;
          f32x4 v = {0.f, 0.f, 0.f, 0.f};
          if (gr < M) {
            long long row = (MODE == 0) ? gr : ga.idx0[gr];
            v = *(const f32x4*)(ga.A + row * ga.lda + kbase + kq);
            if (MODE == 0 && ga.row_scale) {
              float s = ga.row_scale[gr];
#pragma unroll
              for (int c = 0; c < 4; ++c) v[c] *= s;
            }
          }
          store_quad(&As[m * LDSA + kq], v);
        }
      }
    } else {
      // Scalar fallback (x@input_W lda=133; rw K=20; edge-attr tail K-step)
#pragma unroll
      for (int t = 0; t < (MB * KB) / 256; ++t) {
        int e = tid + t * 256;
        int m = e >> 5, k = e & 31;
        int gr = m0 + m;
        float v = (gr < M) ? fetchA<MODE>(ga, gr, kbase + k, K) : 0.f;
        As[m * LDSA + k] = (__bf16)v;
      }
    }

    // ---- B tile 32x64, stored N-major (transposed) ---------------------
    // float4 global loads (ldb = 256, always aligned); transpose needs
    // 4 scalar ds_store_b16 per quad.
#pragma unroll
    for (int t = 0; t < (NBLK * KB) / (256 * 4); ++t) {
      int e4 = tid + t * 256;                // 512 quads
      int k  = e4 >> 4;                      // 16 quads per 64-wide row
      int nq = (e4 & 15) << 2;
      int gk = kbase + k;
      f32x4 v = {0.f, 0.f, 0.f, 0.f};
      if (gk < K) v = *(const f32x4*)(B + (long long)gk * NHID + (n0 + nq));
#pragma unroll
      for (int c = 0; c < 4; ++c) Bs[(nq + c) * LDSB + k] = (__bf16)v[c];
    }
    __syncthreads();

    v16bf afrag = frag_from_lds(&As[(wave * 16) * LDSA], LDSA, lane);
#pragma unroll
    for (int t = 0; t < 4; ++t) {
      v16bf bfrag = frag_from_lds(&Bs[(t * 16) * LDSB], LDSB, lane);
      acc[t] = __builtin_amdgcn_wmma_f32_16x16x32_bf16(
          false, afrag, false, bfrag, (short)0, acc[t], false, false);
    }
  }

  // C/D layout: VGPR j holds M = j + 8*(lane>=16), N = lane&15.
  const int mrow  = m0 + wave * 16 + ((lane >> 4) << 3);
  const int ncol0 = n0 + (lane & 15);
#pragma unroll
  for (int t = 0; t < 4; ++t) {
    const int n  = ncol0 + t * 16;
    const float bv = bias ? bias[n] : 0.f;
#pragma unroll
    for (int j = 0; j < 8; ++j) {
      const int m = mrow + j;
      if (fullM || m < M) {
        float v = acc[t][j] + bv;
        if (flags & 2) v += resid[(long long)m * NHID + n];
        if (flags & 1) v = fmaxf(v, 0.f);
        C[(long long)m * NHID + n] = v;
      }
    }
  }
}

// ---------------------------------------------------------------------------
// Bandwidth-bound helpers (wave-per-row, 256-wide feature rows)
// ---------------------------------------------------------------------------
__global__ void gather_rows(const float* __restrict__ src, const int* __restrict__ map,
                            float* __restrict__ dst, int rows) {
  int r = blockIdx.x * 8 + (threadIdx.x >> 5);
  if (r >= rows) return;
  int lane = threadIdx.x & 31;
  const float* s = src + (long long)map[r] * NHID;
  float* d = dst + (long long)r * NHID;
#pragma unroll
  for (int i = 0; i < 8; ++i) d[lane + i * 32] = s[lane + i * 32];
}

__global__ void gather1d(const float* src, const int* map, float* dst, int n) {
  int i = blockIdx.x * 256 + threadIdx.x;
  if (i < n) dst[i] = src[map[i]];
}

// agg[dst] += (h[src] + e) * ew   (one wave per edge, coalesced 256-f rows)
__global__ void msg_scatter(const float* __restrict__ h, const float* __restrict__ e,
                            const int* __restrict__ src, const int* __restrict__ dst,
                            const float* __restrict__ ew, float* __restrict__ agg, int nE) {
  int ed = blockIdx.x * 8 + (threadIdx.x >> 5);
  if (ed >= nE) return;
  int lane = threadIdx.x & 31;
  const float w = ew[ed];
  const float* hr = h + (long long)src[ed] * NHID;
  const float* er = e + (long long)ed * NHID;
  float* ar = agg + (long long)dst[ed] * NHID;
#pragma unroll
  for (int i = 0; i < 8; ++i) {
    int f = lane + i * 32;
    atomicAdd(&ar[f], (hr[f] + er[f]) * w);
  }
}

__global__ void seg_sum_rows(const float* __restrict__ v, const int* __restrict__ seg,
                             float* __restrict__ sum, int rows) {
  int r = blockIdx.x * 8 + (threadIdx.x >> 5);
  if (r >= rows) return;
  int lane = threadIdx.x & 31;
  const float* vr = v + (long long)r * NHID;
  float* sr = sum + (long long)seg[r] * NHID;
#pragma unroll
  for (int i = 0; i < 8; ++i) atomicAdd(&sr[lane + i * 32], vr[lane + i * 32]);
}

__global__ void seg_sum_rows_gather(const float* __restrict__ v, const int* __restrict__ map,
                                    const int* __restrict__ seg, float* __restrict__ sum,
                                    int rows) {
  int r = blockIdx.x * 8 + (threadIdx.x >> 5);
  if (r >= rows) return;
  int lane = threadIdx.x & 31;
  const float* vr = v + (long long)map[r] * NHID;
  float* sr = sum + (long long)seg[r] * NHID;
#pragma unroll
  for (int i = 0; i < 8; ++i) atomicAdd(&sr[lane + i * 32], vr[lane + i * 32]);
}

__global__ void seg_count(const int* seg, float* cnt, int rows) {
  int i = blockIdx.x * 256 + threadIdx.x;
  if (i < rows) atomicAdd(&cnt[seg[i]], 1.f);
}

__global__ void seg_div(float* sum, const float* cnt, int total) {
  int i = blockIdx.x * 256 + threadIdx.x;
  if (i < total) sum[i] /= fmaxf(cnt[i >> 8], 1.f);
}

__global__ void add_vec(const float* a, const float* b, float* o, int n) {
  int i = blockIdx.x * 256 + threadIdx.x;
  if (i < n) o[i] = a[i] + b[i];
}

__global__ void build_idx(const int* __restrict__ n_patches, const int* __restrict__ ctx_sub,
                          const int* __restrict__ tgt_sub, int* __restrict__ ctx_idx,
                          int* __restrict__ tgt_idx) {
  __shared__ int bi[GB];
  int g = threadIdx.x;
  if (g == 0) {
    int s = 0;
    for (int i = 0; i < GB; ++i) { bi[i] = s; s += n_patches[i]; }
  }
  __syncthreads();
  ctx_idx[g] = ctx_sub[g] + bi[g];
#pragma unroll
  for (int t = 0; t < NTGT; ++t) tgt_idx[g * NTGT + t] = tgt_sub[g * NTGT + t] + bi[g];
}

__global__ void emit_vis_context(const float* __restrict__ emb_sub,
                                 const int* __restrict__ ctx_idx, float* __restrict__ out) {
  int idx = blockIdx.x * 256 + threadIdx.x;      // G*256
  int g = idx >> 8, f = idx & 255;
  out[idx] = emb_sub[(long long)ctx_idx[g] * NHID + f];
}

__global__ void emit_emb_tgt(const float* __restrict__ sff, const int* __restrict__ tgt_idx,
                             float* __restrict__ out_emb, float* __restrict__ out_vis) {
  int idx = blockIdx.x * 256 + threadIdx.x;      // G*NTGT*256
  int r = idx >> 8, f = idx & 255;
  float v = sff[(long long)tgt_idx[r] * NHID + f];
  out_emb[idx] = v;
  if ((r & (NTGT - 1)) == 0) out_vis[(r >> 2) * NHID + f] = v;
}

// cond = emb_ctx[g] + (patch_pe[tgt_idx] @ prw_W + prw_b)   (K=20, VALU is fine)
__global__ void build_cond(const float* __restrict__ emb_sub, const int* __restrict__ ctx_idx,
                           const int* __restrict__ tgt_idx, const float* __restrict__ patch_pe,
                           const float* __restrict__ prw_W, const float* __restrict__ prw_b,
                           float* __restrict__ cond) {
  int idx = blockIdx.x * 256 + threadIdx.x;      // G*NTGT*256
  int r = idx >> 8, f = idx & 255;
  float acc = prw_b[f] + emb_sub[(long long)ctx_idx[r >> 2] * NHID + f];
  const float* pe = patch_pe + (long long)tgt_idx[r] * PRWD;
#pragma unroll
  for (int k = 0; k < PRWD; ++k) acc += pe[k] * prw_W[k * NHID + f];
  cond[idx] = acc;
}

// relu(layernorm(z)) — one wave32 per 256-wide row, shuffle reductions
__global__ void ln_relu(const float* __restrict__ z, const float* __restrict__ g,
                        const float* __restrict__ b, float* __restrict__ out, int rows) {
  int r = blockIdx.x * 8 + (threadIdx.x >> 5);
  if (r >= rows) return;
  int lane = threadIdx.x & 31;
  const float* zr = z + (long long)r * NHID;
  float v[8], s = 0.f;
#pragma unroll
  for (int i = 0; i < 8; ++i) { v[i] = zr[lane + i * 32]; s += v[i]; }
#pragma unroll
  for (int o = 16; o; o >>= 1) s += __shfl_xor(s, o, 32);
  float mu = s * (1.f / NHID);
  float q = 0.f;
#pragma unroll
  for (int i = 0; i < 8; ++i) { float d = v[i] - mu; q += d * d; }
#pragma unroll
  for (int o = 16; o; o >>= 1) q += __shfl_xor(q, o, 32);
  float inv = rsqrtf(q * (1.f / NHID) + 1e-5f);
#pragma unroll
  for (int i = 0; i < 8; ++i) {
    int f = lane + i * 32;
    out[(long long)r * NHID + f] = fmaxf((v[i] - mu) * inv * g[f] + b[f], 0.f);
  }
}

// ---------------------------------------------------------------------------
extern "C" void kernel_launch(void* const* d_in, const int* in_sizes, int n_in,
                              void* d_out, int out_size, void* d_ws, size_t ws_size,
                              hipStream_t stream) {
  (void)in_sizes; (void)n_in; (void)out_size; (void)ws_size;

  const float* x          = (const float*)d_in[0];
  const float* rw_pe      = (const float*)d_in[1];
  const float* node_w     = (const float*)d_in[2];
  const int*   edge_index = (const int*)d_in[3];     // 2 x E
  const float* edge_attr  = (const float*)d_in[4];
  const float* edge_w     = (const float*)d_in[5];
  const int*   batch      = (const int*)d_in[6];
  const int*   sn_map     = (const int*)d_in[7];
  const int*   se_map     = (const int*)d_in[8];
  const int*   comb       = (const int*)d_in[9];     // 2 x ESUB
  const int*   sb         = (const int*)d_in[10];
  const int*   n_patches  = (const int*)d_in[11];
  const int*   ctx_sub    = (const int*)d_in[12];
  const int*   tgt_sub    = (const int*)d_in[13];
  const float* patch_pe   = (const float*)d_in[14];
  // params flattened in dict insertion order
  const float* input_W = (const float*)d_in[15];
  const float* input_b = (const float*)d_in[16];
  const float* rw_W    = (const float*)d_in[17];
  const float* rw_b    = (const float*)d_in[18];
  const float* prw_W   = (const float*)d_in[19];
  const float* prw_b   = (const float*)d_in[20];
  const float* cW_e = (const float*)d_in[21];
  const float* cb_e = (const float*)d_in[22];
  const float* cW   = (const float*)d_in[23];
  const float* cb   = (const float*)d_in[24];
  const float* cW_o = (const float*)d_in[25];
  const float* cb_o = (const float*)d_in[26];
  const float* tW_e = (const float*)d_in[27];
  const float* tb_e = (const float*)d_in[28];
  const float* tW   = (const float*)d_in[29];
  const float* tb   = (const float*)d_in[30];
  const float* tW_o = (const float*)d_in[31];
  const float* tb_o = (const float*)d_in[32];
  const float* pW1  = (const float*)d_in[33];
  const float* pb1  = (const float*)d_in[34];
  const float* ln_g = (const float*)d_in[35];
  const float* ln_b = (const float*)d_in[36];
  const float* pW2  = (const float*)d_in[37];
  const float* pb2  = (const float*)d_in[38];

  // ---- workspace layout (floats) ---------------------------------------
  float* ws = (float*)d_ws;
  size_t o = 0;
  float* base    = ws + o; o += (size_t)NNODE * NHID;   // 25.6 M
  float* bufH    = ws + o; o += (size_t)NSUBN * NHID;   // 51.2 M (ctx h / full_emb)
  float* bufE    = ws + o; o += (size_t)NSUBE * NHID;   // 153.6 M (edge MLP out)
  float* bufAgg  = ws + o; o += (size_t)NSUBN * NHID;   // 51.2 M (agg / hs out)
  float* nw_sub  = ws + o; o += NSUBN;
  float* ew_sub  = ws + o; o += NSUBE;
  float* emb_sub = ws + o; o += (size_t)NPATCH * NHID;
  float* sff     = ws + o; o += (size_t)NPATCH * NHID;  // sub_from_full
  float* cnt_p   = ws + o; o += NPATCH;
  float* cnt_g   = ws + o; o += GB;
  float* bias_in = ws + o; o += NHID;
  float* cond    = ws + o; o += (size_t)GB * NTGT * NHID;
  float* zbuf    = ws + o; o += (size_t)GB * NTGT * NHID;
  float* hln     = ws + o; o += (size_t)GB * NTGT * NHID;
  int*   ctx_idx = (int*)(ws + o); o += GB;
  int*   tgt_idx = (int*)(ws + o); o += (size_t)GB * NTGT;

  // ---- output regions ---------------------------------------------------
  float* out_emb_tgt = (float*)d_out;                       // G*NTGT*NHID
  float* out_pred    = out_emb_tgt + (size_t)GB * NTGT * NHID;
  float* out_visctx  = out_pred    + (size_t)GB * NTGT * NHID;
  float* out_vistgt  = out_visctx  + (size_t)GB * NHID;
  float* out_visgrph = out_vistgt  + (size_t)GB * NHID;

  auto grid_of = [](int M) { return dim3((M + MB - 1) / MB, NHID / NBLK); };

  // 1) bias_in = input_b + rw_b
  //    base = x @ input_W + bias_in          (plain GEMM, K=133, scalar fill)
  //    base += rw_pe @ rw_W                  (plain GEMM, K=20, scalar fill)
  add_vec<<<1, 256, 0, stream>>>(input_b, rw_b, bias_in, NHID);
  {
    GemmA ga{}; ga.A = x; ga.lda = NFN;                       // lda 133: scalar
    wmma_gemm<0><<<grid_of(NNODE), 256, 0, stream>>>(ga, input_W, bias_in, nullptr,
                                                     base, NNODE, NFN, 0);
  }
  {
    GemmA ga{}; ga.A = rw_pe; ga.lda = RWD;                   // K=20 ragged: scalar
    wmma_gemm<0><<<grid_of(NNODE), 256, 0, stream>>>(ga, rw_W, nullptr, base,
                                                     base, NNODE, RWD, 2);
  }

  // 2) gathers for the ctx branch
  gather1d<<<(NSUBN + 255) / 256, 256, 0, stream>>>(node_w, sn_map, nw_sub, NSUBN);
  gather1d<<<(NSUBE + 255) / 256, 256, 0, stream>>>(edge_w, se_map, ew_sub, NSUBE);
  gather_rows<<<(NSUBN + 7) / 8, 256, 0, stream>>>(base, sn_map, bufH, NSUBN);

  // 3) ctx edge MLP: e = relu([h[src] | ea[se_map]] @ W_e + b_e)   (K=270)
  {
    GemmA ga{}; ga.A = bufH; ga.lda = NHID; ga.idx0 = comb;
    ga.idx1 = se_map; ga.ea = edge_attr; ga.K1 = NHID;
    wmma_gemm<2><<<grid_of(NSUBE), 256, 0, stream>>>(ga, cW_e, cb_e, nullptr,
                                                     bufE, NSUBE, NHID + NFE, 1);
  }

  // 4) ctx message-passing layers: h = relu(h + agg @ W[l] + b[l])
  for (int l = 0; l < 3; ++l) {
    hipMemsetAsync(bufAgg, 0, (size_t)NSUBN * NHID * sizeof(float), stream);
    msg_scatter<<<(NSUBE + 7) / 8, 256, 0, stream>>>(bufH, bufE, comb, comb + NSUBE,
                                                     ew_sub, bufAgg, NSUBE);
    GemmA ga{}; ga.A = bufAgg; ga.lda = NHID; ga.vec4 = 1;
    wmma_gemm<0><<<grid_of(NSUBN), 256, 0, stream>>>(
        ga, cW + (size_t)l * NHID * NHID, cb + (size_t)l * NHID, bufH, bufH,
        NSUBN, NHID, 3);
  }
  // 5) ctx out-proj: hs = (h * nw) @ W_o + b_o
  {
    GemmA ga{}; ga.A = bufH; ga.lda = NHID; ga.row_scale = nw_sub; ga.vec4 = 1;
    wmma_gemm<0><<<grid_of(NSUBN), 256, 0, stream>>>(ga, cW_o, cb_o, nullptr,
                                                     bufAgg, NSUBN, NHID, 0);
  }
  // 6) emb_sub = seg_mean(hs, sb, NPATCH)
  hipMemsetAsync(emb_sub, 0, (size_t)NPATCH * NHID * sizeof(float), stream);
  hipMemsetAsync(cnt_p, 0, NPATCH * sizeof(float), stream);
  seg_sum_rows<<<(NSUBN + 7) / 8, 256, 0, stream>>>(bufAgg, sb, emb_sub, NSUBN);
  seg_count<<<(NSUBN + 255) / 256, 256, 0, stream>>>(sb, cnt_p, NSUBN);
  seg_div<<<(NPATCH * NHID) / 256, 256, 0, stream>>>(emb_sub, cnt_p, NPATCH * NHID);

  // 7) tgt edge MLP on the full graph (h0 = base)
  {
    GemmA ga{}; ga.A = base; ga.lda = NHID; ga.idx0 = edge_index;
    ga.ea = edge_attr; ga.K1 = NHID;
    wmma_gemm<3><<<grid_of(NEDGE), 256, 0, stream>>>(ga, tW_e, tb_e, nullptr,
                                                     bufE, NEDGE, NHID + NFE, 1);
  }
  // 8) tgt layers (h = base, updated in place)
  for (int l = 0; l < 3; ++l) {
    hipMemsetAsync(bufAgg, 0, (size_t)NNODE * NHID * sizeof(float), stream);
    msg_scatter<<<(NEDGE + 7) / 8, 256, 0, stream>>>(base, bufE, edge_index,
                                                     edge_index + NEDGE, edge_w,
                                                     bufAgg, NEDGE);
    GemmA ga{}; ga.A = bufAgg; ga.lda = NHID; ga.vec4 = 1;
    wmma_gemm<0><<<grid_of(NNODE), 256, 0, stream>>>(
        ga, tW + (size_t)l * NHID * NHID, tb + (size_t)l * NHID, base, base,
        NNODE, NHID, 3);
  }
  // 9) full_emb = (h * node_w) @ W_o + b_o   -> bufH
  {
    GemmA ga{}; ga.A = base; ga.lda = NHID; ga.row_scale = node_w; ga.vec4 = 1;
    wmma_gemm<0><<<grid_of(NNODE), 256, 0, stream>>>(ga, tW_o, tb_o, nullptr,
                                                     bufH, NNODE, NHID, 0);
  }
  // 10) vis_graph = seg_mean(full_emb, batch, G)  (accumulate in d_out region)
  hipMemsetAsync(out_visgrph, 0, (size_t)GB * NHID * sizeof(float), stream);
  hipMemsetAsync(cnt_g, 0, GB * sizeof(float), stream);
  seg_sum_rows<<<(NNODE + 7) / 8, 256, 0, stream>>>(bufH, batch, out_visgrph, NNODE);
  seg_count<<<(NNODE + 255) / 256, 256, 0, stream>>>(batch, cnt_g, NNODE);
  seg_div<<<(GB * NHID) / 256, 256, 0, stream>>>(out_visgrph, cnt_g, GB * NHID);

  // 11) sub_from_full = seg_mean(full_emb[sn_map], sb, NPATCH)
  hipMemsetAsync(sff, 0, (size_t)NPATCH * NHID * sizeof(float), stream);
  hipMemsetAsync(cnt_p, 0, NPATCH * sizeof(float), stream);
  seg_sum_rows_gather<<<(NSUBN + 7) / 8, 256, 0, stream>>>(bufH, sn_map, sb, sff, NSUBN);
  seg_count<<<(NSUBN + 255) / 256, 256, 0, stream>>>(sb, cnt_p, NSUBN);
  seg_div<<<(NPATCH * NHID) / 256, 256, 0, stream>>>(sff, cnt_p, NPATCH * NHID);

  // 12) indices + output emits
  build_idx<<<1, GB, 0, stream>>>(n_patches, ctx_sub, tgt_sub, ctx_idx, tgt_idx);
  emit_vis_context<<<GB, 256, 0, stream>>>(emb_sub, ctx_idx, out_visctx);
  emit_emb_tgt<<<GB * NTGT, 256, 0, stream>>>(sff, tgt_idx, out_emb_tgt, out_vistgt);

  // 13) predictor: cond = emb_ctx + tpes ; pred = relu(ln(cond@W1+b1))@W2+b2
  build_cond<<<GB * NTGT, 256, 0, stream>>>(emb_sub, ctx_idx, tgt_idx, patch_pe,
                                            prw_W, prw_b, cond);
  {
    GemmA ga{}; ga.A = cond; ga.lda = NHID; ga.vec4 = 1;
    wmma_gemm<0><<<grid_of(GB * NTGT), 256, 0, stream>>>(ga, pW1, pb1, nullptr,
                                                         zbuf, GB * NTGT, NHID, 0);
  }
  ln_relu<<<(GB * NTGT + 7) / 8, 256, 0, stream>>>(zbuf, ln_g, ln_b, hln, GB * NTGT);
  {
    GemmA ga{}; ga.A = hln; ga.lda = NHID; ga.vec4 = 1;
    wmma_gemm<0><<<grid_of(GB * NTGT), 256, 0, stream>>>(ga, pW2, pb2, nullptr,
                                                         out_pred, GB * NTGT, NHID, 0);
  }
}
